// RnnModel_10711648436745
// MI455X (gfx1250) — matrix operand order repaired and checked by reading, not compiled, not run on hardware
//
#include <hip/hip_runtime.h>
#include <math.h>

#define VOCAB  8000
#define HIDDEN 256
#define BATCH  32
#define SEQ    256
#define HSTRIDE 264   // padded LDS row stride (halves): 264*2B=528B -> 132 dw, 132%64=4 -> conflict-free b128

typedef __attribute__((ext_vector_type(16))) _Float16 v16h;
typedef __attribute__((ext_vector_type(8)))  _Float16 v8h;
typedef __attribute__((ext_vector_type(8)))  float    v8f;

union V16 { v16h v; v8h h[2]; };

// Hardware tanh (gfx1250 V_TANH_F32 trans op) instead of branchy libm expansion.
__device__ __forceinline__ float hw_tanh(float x) {
#if __has_builtin(__builtin_amdgcn_tanhf)
    return __builtin_amdgcn_tanhf(x);
#else
    float r;
    asm volatile("v_tanh_f32 %0, %1" : "=v"(r) : "v"(x));
    return r;
#endif
}

// ---------------------------------------------------------------------------
// emb[(t*B+b)*H + n] = W_ih[n, X[b,t]] + b_ih[n] + b_hh[n]   (biases folded in)
// ---------------------------------------------------------------------------
__global__ void embed_kernel(const float* __restrict__ W_ih,
                             const float* __restrict__ b_ih,
                             const float* __restrict__ b_hh,
                             const int*   __restrict__ X,
                             float*       __restrict__ emb)
{
    int idx = blockIdx.x * blockDim.x + threadIdx.x;   // 0 .. SEQ*BATCH*HIDDEN-1
    int n = idx & (HIDDEN - 1);
    int r = idx >> 8;            // t*BATCH + b
    int t = r >> 5;
    int b = r & 31;
    int x = X[b * SEQ + t];
    emb[idx] = W_ih[(size_t)n * VOCAB + x] + b_ih[n] + b_hh[n];
}

__global__ void cvt_f32_f16(const float* __restrict__ src, _Float16* __restrict__ dst, int n)
{
    int i = blockIdx.x * blockDim.x + threadIdx.x;
    if (i < n) dst[i] = (_Float16)src[i];
}

// ---------------------------------------------------------------------------
// Persistent single-WGP recurrence. 512 threads = 16 waves.
// Wave tile: 16(M=batch) x 32(N=hidden) -> 2 WMMA tiles; waves: 2(M) x 8(N).
// W_hh fragments (16 x v16h per lane) preloaded in VGPRs; h ping-pongs in LDS (f16).
// ---------------------------------------------------------------------------
__global__ void __launch_bounds__(512, 1)
rnn_kernel(const float*    __restrict__ emb,
           const _Float16* __restrict__ Whh16,
           const float*    __restrict__ h0,
           _Float16*       __restrict__ Y16,
           float*          __restrict__ h_last)
{
    __shared__ _Float16 hbuf[2][BATCH * HSTRIDE];

    const int tid   = threadIdx.x;
    const int lane  = tid & 31;
    const int wave  = tid >> 5;          // 0..15
    const int m0    = (wave & 1) * 16;   // batch tile
    const int n0    = (wave >> 1) * 32;  // hidden-out group (2 tiles of 16)
    const int lmod  = lane & 15;
    const int lhalf = lane >> 4;

    // init h from h0 (f32 -> f16 into LDS)
    for (int i = tid; i < BATCH * HIDDEN; i += 512) {
        int b = i >> 8, n = i & 255;
        hbuf[0][b * HSTRIDE + n] = (_Float16)h0[i];
    }

    // Preload B fragments: B = W_hh^T, so B column n == W_hh row n (K-contiguous).
    // B layout: lane holds K {+0..7} and {+16..23} (hi lanes: +8.. / +24..) within each K=32 chunk.
    v16h bfrag[2][8];
#pragma unroll
    for (int i = 0; i < 2; ++i) {
        const _Float16* wr = Whh16 + (size_t)(n0 + 16 * i + lmod) * HIDDEN;
#pragma unroll
        for (int kc = 0; kc < 8; ++kc)
        {
            V16 bb;
            bb.h[0] = *(const v8h*)(wr + kc * 32 + lhalf * 8);
            bb.h[1] = *(const v8h*)(wr + kc * 32 + 16 + lhalf * 8);
            bfrag[i][kc] = bb.v;
        }
    }
    __syncthreads();

    const int mb = m0 + lhalf * 8;   // C/D layout: VGPR r -> M = mb + r, N = lmod

    for (int t = 0; t < SEQ; ++t) {
        const _Float16* hc = hbuf[t & 1];
        _Float16*       hn = hbuf[(t + 1) & 1];

        // Prefetch this step's additive input (emb + biases) so the global loads
        // overlap the 16 WMMAs below.
        float e0[8], e1[8];
#pragma unroll
        for (int r = 0; r < 8; ++r) {
            const size_t rowoff = ((size_t)t * BATCH + (mb + r)) * HIDDEN;
            e0[r] = emb[rowoff + n0 + lmod];
            e1[r] = emb[rowoff + n0 + 16 + lmod];
        }

        v8f acc0 = {}, acc1 = {};
        const _Float16* arow = hc + (m0 + lmod) * HSTRIDE;
#pragma unroll
        for (int kc = 0; kc < 8; ++kc) {
            // A layout (16x32 f16): lane holds K {+0..7} and {+16..23} (hi lanes: +8.. / +24..)
            V16 a;
            a.h[0] = *(const v8h*)(arow + kc * 32 + lhalf * 8);
            a.h[1] = *(const v8h*)(arow + kc * 32 + 16 + lhalf * 8);
            acc0 = __builtin_amdgcn_wmma_f32_16x16x32_f16(false, a.v, false, bfrag[0][kc],
                                                          (short)0, acc0, false, false);
            acc1 = __builtin_amdgcn_wmma_f32_16x16x32_f16(false, a.v, false, bfrag[1][kc],
                                                          (short)0, acc1, false, false);
        }

        // epilogue: h_new = tanh(acc + emb_t)  -- single v_tanh_f32 per element
#pragma unroll
        for (int r = 0; r < 8; ++r) {
            const int b = mb + r;
            const size_t rowoff = ((size_t)t * BATCH + b) * HIDDEN;
            {
                const int n = n0 + lmod;
                float v = hw_tanh(acc0[r] + e0[r]);
                _Float16 vh = (_Float16)v;
                hn[b * HSTRIDE + n] = vh;
                Y16[rowoff + n] = vh;
                if (t == SEQ - 1) h_last[b * HIDDEN + n] = v;
            }
            {
                const int n = n0 + 16 + lmod;
                float v = hw_tanh(acc1[r] + e1[r]);
                _Float16 vh = (_Float16)v;
                hn[b * HSTRIDE + n] = vh;
                Y16[rowoff + n] = vh;
                if (t == SEQ - 1) h_last[b * HIDDEN + n] = v;
            }
        }
        __syncthreads();   // hn fully written before next step reads it
    }
}

// ---------------------------------------------------------------------------
// out(8192x8000) = Y(8192x256) @ W_out^T + b_out.  Block = 8 waves, tile 128M x 64N.
// Wave tile 16M x 64N (4 WMMA tiles), K=256 in 8 chunks. A/B f16 from global (L2-resident).
// ---------------------------------------------------------------------------
__global__ void __launch_bounds__(256, 2)
gemm_kernel(const _Float16* __restrict__ Y16,
            const _Float16* __restrict__ Wout16,
            const float*    __restrict__ b_out,
            float*          __restrict__ out)
{
    const int tid   = threadIdx.x;
    const int lane  = tid & 31;
    const int wave  = tid >> 5;   // 0..7
    const int lmod  = lane & 15;
    const int lhalf = lane >> 4;

    const int m0 = blockIdx.y * 128 + wave * 16;
    const int n0 = blockIdx.x * 64;

    v8f acc[4] = {v8f{}, v8f{}, v8f{}, v8f{}};

    const _Float16* arow = Y16 + (size_t)(m0 + lmod) * HIDDEN;
#pragma unroll
    for (int kc = 0; kc < 8; ++kc) {
        V16 a;
        a.h[0] = *(const v8h*)(arow + kc * 32 + lhalf * 8);
        a.h[1] = *(const v8h*)(arow + kc * 32 + 16 + lhalf * 8);
#pragma unroll
        for (int i = 0; i < 4; ++i) {
            const _Float16* brow = Wout16 + (size_t)(n0 + 16 * i + lmod) * HIDDEN;
            V16 bb;
            bb.h[0] = *(const v8h*)(brow + kc * 32 + lhalf * 8);
            bb.h[1] = *(const v8h*)(brow + kc * 32 + 16 + lhalf * 8);
            v16h b = bb.v;
            acc[i] = __builtin_amdgcn_wmma_f32_16x16x32_f16(false, a.v, false, b,
                                                            (short)0, acc[i], false, false);
        }
    }

    const int mb = m0 + lhalf * 8;
#pragma unroll
    for (int i = 0; i < 4; ++i) {
        const int n = n0 + 16 * i + lmod;
        const float bias = b_out[n];
#pragma unroll
        for (int r = 0; r < 8; ++r)
            out[(size_t)(mb + r) * VOCAB + n] = acc[i][r] + bias;
    }
}

// ---------------------------------------------------------------------------
extern "C" void kernel_launch(void* const* d_in, const int* in_sizes, int n_in,
                              void* d_out, int out_size, void* d_ws, size_t ws_size,
                              hipStream_t stream)
{
    const int*   X     = (const int*)  d_in[0];
    const float* h0    = (const float*)d_in[1];
    const float* W_ih  = (const float*)d_in[2];
    const float* b_ih  = (const float*)d_in[3];
    const float* W_hh  = (const float*)d_in[4];
    const float* b_hh  = (const float*)d_in[5];
    const float* W_out = (const float*)d_in[6];
    const float* b_out = (const float*)d_in[7];
    float* out = (float*)d_out;

    char* ws = (char*)d_ws;
    float*    emb    = (float*)ws;                                      // 8,388,608 B
    _Float16* Y16    = (_Float16*)(ws + 8388608);                       // 4,194,304 B
    _Float16* Wout16 = (_Float16*)(ws + 8388608 + 4194304);             // 4,096,000 B
    _Float16* Whh16  = (_Float16*)(ws + 8388608 + 4194304 + 4096000);   //   131,072 B

    embed_kernel<<<(SEQ * BATCH * HIDDEN) / 256, 256, 0, stream>>>(W_ih, b_ih, b_hh, X, emb);
    cvt_f32_f16<<<(VOCAB * HIDDEN + 255) / 256, 256, 0, stream>>>(W_out, Wout16, VOCAB * HIDDEN);
    cvt_f32_f16<<<(HIDDEN * HIDDEN + 255) / 256, 256, 0, stream>>>(W_hh, Whh16, HIDDEN * HIDDEN);

    float* h_last = out + (size_t)out_size - BATCH * HIDDEN;   // tuple tail: h_last (1,32,256) f32
    rnn_kernel<<<1, 512, 0, stream>>>(emb, Whh16, h0, Y16, h_last);

    gemm_kernel<<<dim3(VOCAB / 64, (SEQ * BATCH) / 128), 256, 0, stream>>>(Y16, Wout16, b_out, out);
}
